// GraphCast_38139309589240
// MI455X (gfx1250) — matrix operand order repaired and checked by reading, not compiled, not run on hardware
//
#include <hip/hip_runtime.h>
#include <hip/hip_bf16.h>

// ---------------------------------------------------------------------------
// GraphCast-style GNN for MI455X (gfx1250, wave32, WMMA).
// All dense math runs through v_wmma_f32_16x16x32_f16.
// ---------------------------------------------------------------------------

typedef _Float16 half_t;
typedef __attribute__((ext_vector_type(16))) _Float16 v16h;
typedef __attribute__((ext_vector_type(8)))  float    v8f;

#define BLOCK  256
#define NWAVES 8
#define MAX_K  384   // max input / hidden width

struct MlpArgs {
  // input = concat(seg0, seg1, seg2); base==nullptr -> zeros; idx==nullptr -> row
  const float* s0b; const int* s0i; int s0w;
  const float* s1b; const int* s1i; int s1w;
  const float* s2b; const int* s2i; int s2w;
  int nrows;
  const half_t* Wt1; const float* b1; const float* g1; const float* be1; // hidden: W^T [Op1,K1p] f16
  int K1p, O1, K2p;
  const half_t* Wt2; const float* b2; const float* g2; const float* be2; // out: g2==nullptr -> no LN
  int O2;
  float* out;
  const float* residual;  // out += residual (after optional mask)
  const float* deg;       // if set: out *= (deg[row] > 0)
};

// A fragment (16x32 f16, CDNA5 layout): lane L<16 -> row L, K {0..7,16..23};
// lane L>=16 -> row L-16, K {8..15,24..31}.
__device__ __forceinline__ v16h load_a_frag(const _Float16* base, int strideH, int lane, int kk) {
  int m = lane & 15;
  const _Float16* p = base + m * strideH + kk + ((lane >= 16) ? 8 : 0);
  v16h a;
#pragma unroll
  for (int e = 0; e < 8; ++e) a[e] = p[e];
#pragma unroll
  for (int e = 8; e < 16; ++e) a[e] = p[e + 8];
  return a;
}

// One 16x16 output tile: full K sweep for tile t.
// B fragment (32x16 f16): lane holds column n = t*16 + (lane&15); lanes 0-15
// K {0..7,16..23}, lanes 16-31 K {8..15,24..31}. Wt is transposed [O, Kp].
__device__ __forceinline__ v8f gemm_tile(const _Float16* sA, const half_t* Wt,
                                         int Kp, int t, int lane) {
  v8f acc;
#pragma unroll
  for (int r = 0; r < 8; ++r) acc[r] = 0.f;
  const half_t* wp = Wt + (long long)(t * 16 + (lane & 15)) * Kp + ((lane >= 16) ? 8 : 0);
  for (int kk = 0; kk < Kp; kk += 32) {
    v16h a = load_a_frag(sA, Kp, lane, kk);
    typedef __attribute__((ext_vector_type(8))) _Float16 v8h_;
    v8h_ blo = *(const v8h_*)(wp + kk);
    v8h_ bhi = *(const v8h_*)(wp + kk + 16);
    v16h b = __builtin_shufflevector(blo, bhi, 0,1,2,3,4,5,6,7,8,9,10,11,12,13,14,15);
    acc = __builtin_amdgcn_wmma_f32_16x16x32_f16(false, a, false, b, (short)0, acc, false, false);
  }
  return acc;
}

// write one 16x16 C tile (+bias) into the LDS row-block buffer
__device__ __forceinline__ void store_tile(float* sH, int W, const float* bias,
                                           const v8f& acc, int t, int lane) {
  int col = t * 16 + (lane & 15);
  if (col < W) {
    int rb = (lane < 16) ? 0 : 8;   // C layout: VGPR r -> row r (lanes<16) / r+8
#pragma unroll
    for (int r = 0; r < 8; ++r)
      sH[(rb + r) * W + col] = acc[r] + bias[col];
  }
}

// row-wise mean/rstd over sH[16][W] using all 256 threads (16 threads per row)
__device__ __forceinline__ void row_stats(const float* sH, int W, float* sRed, float* sStat, int tid) {
  int m = tid >> 4, sub = tid & 15;
  float s = 0.f, q = 0.f;
  for (int c = sub; c < W; c += 16) {
    float v = sH[m * W + c];
    s += v; q += v * v;
  }
  sRed[(m * 16 + sub) * 2]     = s;
  sRed[(m * 16 + sub) * 2 + 1] = q;
  __syncthreads();
  if (sub == 0) {
    float S = 0.f, Q = 0.f;
    for (int j = 0; j < 16; ++j) { S += sRed[(m * 16 + j) * 2]; Q += sRed[(m * 16 + j) * 2 + 1]; }
    float mean = S / W;
    float var  = Q / W - mean * mean;
    sStat[m * 2]     = mean;
    sStat[m * 2 + 1] = rsqrtf(var + 1e-5f);
  }
  __syncthreads();
}

// Fused MLP: out = [LN](silu(LN(gather(x) @ W1 + b1)) @ W2 + b2) [*mask] [+res]
__global__ __launch_bounds__(BLOCK) void k_mlp(MlpArgs A) {
  // sX: first holds the gathered f16 input row-block [16 x K1p] (stage 1 A),
  //     then is overwritten with the f16 hidden activations [16 x K2p] (stage 2 A).
  __shared__ _Float16 sX[16 * MAX_K];
  __shared__ float    sH[16 * MAX_K];     // f32 accum row-block
  __shared__ float    sRed[16 * 16 * 2];
  __shared__ float    sStat[16 * 2];

  const int tid  = threadIdx.x;
  const int lane = tid & 31;
  const int wave = tid >> 5;
  const int r0   = blockIdx.x * 16;

  // ---------------- gather: stage the whole input row-block once ----------
  {
    const int m = tid >> 4, sub = tid & 15;     // 16 threads per row
    const int row = r0 + m;
    const float* p0 = nullptr; const float* p1 = nullptr; const float* p2 = nullptr;
    if (row < A.nrows) {
      if (A.s0b) p0 = A.s0b + (long long)(A.s0i ? A.s0i[row] : row) * A.s0w;
      if (A.s1b) p1 = A.s1b + (long long)(A.s1i ? A.s1i[row] : row) * A.s1w;
      if (A.s2b) p2 = A.s2b + (long long)(A.s2i ? A.s2i[row] : row) * A.s2w;
    }
    const int w01 = A.s0w + A.s1w, w012 = w01 + A.s2w;
    for (int k = sub; k < A.K1p; k += 16) {
      float v = 0.f;
      if (k < A.s0w)     { if (p0) v = p0[k]; }
      else if (k < w01)  { if (p1) v = p1[k - A.s0w]; }
      else if (k < w012) { if (p2) v = p2[k - w01]; }
      sX[m * A.K1p + k] = (_Float16)v;
    }
  }
  __syncthreads();

  // ----------------------- stage 1: hidden layer --------------------------
  // tile-outer / K-inner: guards evaluated once per tile, K loop is a pure
  // ds_load + global_load + wmma stream with constant-stride weight pointer.
  const int nT1 = (A.O1 + 15) >> 4;       // column tiles (<= 24, <= 3 per wave)
  const int t0 = wave, t1 = wave + NWAVES, t2 = wave + 2 * NWAVES;
  const bool u0 = t0 < nT1, u1 = t1 < nT1, u2 = t2 < nT1;
  v8f acc0, acc1, acc2;
#pragma unroll
  for (int r = 0; r < 8; ++r) { acc0[r] = 0.f; acc1[r] = 0.f; acc2[r] = 0.f; }

  if (u0) acc0 = gemm_tile(sX, A.Wt1, A.K1p, t0, lane);
  if (u1) acc1 = gemm_tile(sX, A.Wt1, A.K1p, t1, lane);
  if (u2) acc2 = gemm_tile(sX, A.Wt1, A.K1p, t2, lane);

  if (u0) store_tile(sH, A.O1, A.b1, acc0, t0, lane);
  if (u1) store_tile(sH, A.O1, A.b1, acc1, t1, lane);
  if (u2) store_tile(sH, A.O1, A.b1, acc2, t2, lane);
  __syncthreads();                         // all waves done with sX + sH written
  row_stats(sH, A.O1, sRed, sStat, tid);
  for (int i = tid; i < 16 * A.K2p; i += BLOCK) {  // LN + SiLU -> f16 into sX
    int m = i / A.K2p, c = i - m * A.K2p;
    _Float16 h = (_Float16)0.f;
    if (c < A.O1) {
      float x = sH[m * A.O1 + c];
      x = (x - sStat[m * 2]) * sStat[m * 2 + 1] * A.g1[c] + A.be1[c];
      x = x * (1.f / (1.f + __expf(-x)));          // silu
      h = (_Float16)x;
    }
    sX[i] = h;
  }
  __syncthreads();

  // ----------------------- stage 2: output layer --------------------------
  const int nT2 = (A.O2 + 15) >> 4;   // <= 8 -> one tile per wave
  v8f accO;
#pragma unroll
  for (int r = 0; r < 8; ++r) accO[r] = 0.f;
  if (wave < nT2) accO = gemm_tile(sX, A.Wt2, A.K2p, wave, lane);
  __syncthreads();
  if (wave < nT2) store_tile(sH, A.O2, A.b2, accO, wave, lane);
  __syncthreads();
  if (A.g2) row_stats(sH, A.O2, sRed, sStat, tid);
  for (int i = tid; i < 16 * A.O2; i += BLOCK) {
    int m = i / A.O2, c = i - m * A.O2;
    int row = r0 + m;
    if (row >= A.nrows) continue;
    float x = sH[i];
    if (A.g2) x = (x - sStat[m * 2]) * sStat[m * 2 + 1] * A.g2[c] + A.be2[c];
    if (A.deg) x = (A.deg[row] > 0.f) ? x : 0.f;
    if (A.residual) x += A.residual[(long long)row * A.O2 + c];
    A.out[(long long)row * A.O2 + c] = x;
  }
}

// fp32 [K,O] -> f16 transposed+padded [Op,Kp]
__global__ void k_convert(const float* __restrict__ W, half_t* __restrict__ Wt,
                          int K, int O, int Kp, int Op) {
  int i = blockIdx.x * 256 + threadIdx.x;
  if (i >= Op * Kp) return;
  int o = i / Kp, k = i - o * Kp;
  float v = (o < O && k < K) ? W[(long long)k * O + o] : 0.f;
  Wt[i] = (half_t)v;
}

__global__ void k_scatter_add(const float* __restrict__ e, const int* __restrict__ dst,
                              long long n, float* __restrict__ msum) {
  long long t = (long long)blockIdx.x * 256 + threadIdx.x;
  if (t >= n) return;
  int ei = (int)(t >> 7), c = (int)(t & 127);   // D = 128
  atomicAdd(&msum[(long long)dst[ei] * 128 + c], e[t]);
}

__global__ void k_deg(const int* __restrict__ dst, int E, float* __restrict__ deg) {
  int t = blockIdx.x * 256 + threadIdx.x;
  if (t < E) atomicAdd(&deg[dst[t]], 1.0f);
}

__global__ void k_add_inplace(float* __restrict__ a, const float* __restrict__ b, long long n) {
  long long t = (long long)blockIdx.x * 256 + threadIdx.x;
  if (t < n) a[t] += b[t];
}

// ---------------------------------------------------------------------------
extern "C" void kernel_launch(void* const* d_in, const int* in_sizes, int n_in,
                              void* d_out, int out_size, void* d_ws, size_t ws_size,
                              hipStream_t stream) {
  (void)in_sizes; (void)n_in; (void)out_size;
  constexpr int NG = 65536, NM = 10242, EG2M = 196608, EM2M = 40960, EM2G = 196608, NL = 16;

  const float* weather     = (const float*)d_in[0];
  const float* grid_static = (const float*)d_in[1];
  const float* mesh_static = (const float*)d_in[2];
  const float* g2m_ef      = (const float*)d_in[3];
  const float* m2m_ef      = (const float*)d_in[4];
  const float* m2g_ef      = (const float*)d_in[5];
  const int* g2m_src = (const int*)d_in[6];
  const int* g2m_dst = (const int*)d_in[7];
  const int* m2m_src = (const int*)d_in[8];
  const int* m2m_dst = (const int*)d_in[9];
  const int* m2g_src = (const int*)d_in[10];
  const int* m2g_dst = (const int*)d_in[11];

  // ---- workspace carve-up ----
  size_t off = 0;
  char* wsb = (char*)d_ws;
  auto walloc = [&](size_t bytes) -> void* {
    void* p = wsb + off;
    off += (bytes + 255) & ~(size_t)255;
    return p;
  };
  float* h_grid = (float*)walloc((size_t)NG * 128 * 4);
  float* h_mesh = (float*)walloc((size_t)NM * 128 * 4);
  float* e_g2m  = (float*)walloc((size_t)EG2M * 128 * 4);
  float* e_m2m  = (float*)walloc((size_t)EM2M * 128 * 4);
  float* e_m2g  = (float*)walloc((size_t)EM2G * 128 * 4);
  float* e_new  = (float*)walloc((size_t)EG2M * 128 * 4);
  float* msum   = (float*)walloc((size_t)NG * 128 * 4);
  float* degb   = (float*)walloc((size_t)NG * 4);
  half_t* wtArena = (half_t*)walloc((size_t)16 * 1024 * 1024);
  if (off > ws_size) return;

  // ---- param leaf indexing (jax pytree: dict keys sorted, None dropped) ----
  // mlp(ln): [hW,hb,hg,hbe, lng,lnbe, oW,ob]  mlp(no ln): [hW,hb,hg,hbe, oW,ob]
  // gnn: dst(8), edge(8), src(8)
  // order: g2m(12), g2m_emb(36), grid_emb(44), m2g(52), m2g_emb(76),
  //        m2g_out(84, no-ln), m2m[16](90+24i), m2m_emb(474), mesh_emb(482)
  struct MlpW {
    const float *hW, *hb, *hg, *hbe, *lng, *lnbe, *oW, *ob;
    half_t *Wt1, *Wt2;
    int din, dout, K1p, K2p, Op1, Op2;
  };
  MlpW mlps[64];
  int nm = 0;
  size_t wtOff = 0;
  auto rup  = [](int x, int a) { return (x + a - 1) / a * a; };
  auto leaf = [&](int i) { return (const float*)d_in[i]; };
  auto mkmlp = [&](int base, int din, int dout, bool ln) -> int {
    MlpW& m = mlps[nm];
    m.hW = leaf(base); m.hb = leaf(base + 1); m.hg = leaf(base + 2); m.hbe = leaf(base + 3);
    if (ln) { m.lng = leaf(base + 4); m.lnbe = leaf(base + 5); m.oW = leaf(base + 6); m.ob = leaf(base + 7); }
    else    { m.lng = nullptr;        m.lnbe = nullptr;        m.oW = leaf(base + 4); m.ob = leaf(base + 5); }
    m.din = din; m.dout = dout;
    m.K1p = rup(din, 32); m.K2p = rup(din, 32);
    m.Op1 = rup(din, 16); m.Op2 = rup(dout, 16);
    m.Wt1 = wtArena + wtOff; wtOff += (size_t)m.Op1 * m.K1p;
    m.Wt2 = wtArena + wtOff; wtOff += (size_t)m.Op2 * m.K2p;
    return nm++;
  };

  int g2m_dstM  = mkmlp(12, 256, 128, true);
  int g2m_edgeM = mkmlp(20, 384, 128, true);
  int g2m_srcM  = mkmlp(28, 128, 128, true);
  int g2m_embM  = mkmlp(36,   4, 128, true);
  int grid_embM = mkmlp(44,  64, 128, true);
  int m2g_dstM  = mkmlp(52, 256, 128, true);
  int m2g_edgeM = mkmlp(60, 384, 128, true);
  int m2g_srcM  = mkmlp(68, 128, 128, true);  (void)m2g_srcM; // unused by output
  int m2g_embM  = mkmlp(76,   4, 128, true);
  int m2g_outM  = mkmlp(84, 128,  32, false);
  int m2m_dstM[NL], m2m_edgeM[NL], m2m_srcM[NL];
  for (int l = 0; l < NL; ++l) {
    int base = 90 + 24 * l;
    m2m_dstM[l]  = mkmlp(base,      256, 128, true);
    m2m_edgeM[l] = mkmlp(base + 8,  384, 128, true);
    m2m_srcM[l]  = mkmlp(base + 16, 128, 128, true);
  }
  int m2m_embM  = mkmlp(474,  4, 128, true);
  int mesh_embM = mkmlp(482, 64, 128, true);

  // ---- weight conversion (fp32 -> transposed, padded f16) ----
  for (int i = 0; i < nm; ++i) {
    const MlpW& m = mlps[i];
    int n1 = m.Op1 * m.K1p;
    k_convert<<<(n1 + 255) / 256, 256, 0, stream>>>(m.hW, m.Wt1, m.din, m.din,  m.K1p, m.Op1);
    int n2 = m.Op2 * m.K2p;
    k_convert<<<(n2 + 255) / 256, 256, 0, stream>>>(m.oW, m.Wt2, m.din, m.dout, m.K2p, m.Op2);
  }

  auto run_mlp = [&](int mi,
                     const float* s0b, const int* s0i, int s0w,
                     const float* s1b, const int* s1i, int s1w,
                     const float* s2b, const int* s2i, int s2w,
                     int nrows, float* out, const float* residual, const float* deg) {
    const MlpW& m = mlps[mi];
    MlpArgs A;
    A.s0b = s0b; A.s0i = s0i; A.s0w = s0w;
    A.s1b = s1b; A.s1i = s1i; A.s1w = s1w;
    A.s2b = s2b; A.s2i = s2i; A.s2w = s2w;
    A.nrows = nrows;
    A.Wt1 = m.Wt1; A.b1 = m.hb; A.g1 = m.hg; A.be1 = m.hbe;
    A.K1p = m.K1p; A.O1 = m.din; A.K2p = m.K2p;
    A.Wt2 = m.Wt2; A.b2 = m.ob; A.g2 = m.lng; A.be2 = m.lnbe;
    A.O2 = m.dout;
    A.out = out; A.residual = residual; A.deg = deg;
    k_mlp<<<(nrows + 15) / 16, BLOCK, 0, stream>>>(A);
  };

  auto gnn_step = [&](int dstM, int edgeM, int srcM,
                      float* e_h, float* h_src, float* h_dst,
                      const int* src_idx, const int* dst_idx,
                      int E, int Nd, int Ns, bool do_src, bool do_eadd) {
    // e_new = edgeMLP(concat(e_h, h_src[src], h_dst[dst]))
    run_mlp(edgeM, e_h, nullptr, 128, h_src, src_idx, 128, h_dst, dst_idx, 128,
            E, e_new, nullptr, nullptr);
    hipMemsetAsync(msum, 0, (size_t)Nd * 128 * 4, stream);
    hipMemsetAsync(degb, 0, (size_t)Nd * 4, stream);
    long long n = (long long)E * 128;
    k_scatter_add<<<(int)((n + 255) / 256), 256, 0, stream>>>(e_new, dst_idx, n, msum);
    k_deg<<<(E + 255) / 256, 256, 0, stream>>>(dst_idx, E, degb);
    // h_dst += where(deg>0, dstMLP(concat(h_dst, msum)), 0)
    run_mlp(dstM, h_dst, nullptr, 128, msum, nullptr, 128, nullptr, nullptr, 0,
            Nd, h_dst, h_dst, degb);
    if (do_src)  // h_src += srcMLP(h_src)   (m2m: sees dst-updated h, matching ref)
      run_mlp(srcM, h_src, nullptr, 128, nullptr, nullptr, 0, nullptr, nullptr, 0,
              Ns, h_src, h_src, nullptr);
    if (do_eadd)
      k_add_inplace<<<(int)((n + 255) / 256), 256, 0, stream>>>(e_h, e_new, n);
  };

  // ---- embeddings ----
  run_mlp(grid_embM, weather, nullptr, 32, grid_static, nullptr, 32, nullptr, nullptr, 0,
          NG, h_grid, nullptr, nullptr);
  run_mlp(mesh_embM, nullptr, nullptr, 32, mesh_static, nullptr, 32, nullptr, nullptr, 0,
          NM, h_mesh, nullptr, nullptr);
  run_mlp(g2m_embM, g2m_ef, nullptr, 4, nullptr, nullptr, 0, nullptr, nullptr, 0,
          EG2M, e_g2m, nullptr, nullptr);
  run_mlp(m2m_embM, m2m_ef, nullptr, 4, nullptr, nullptr, 0, nullptr, nullptr, 0,
          EM2M, e_m2m, nullptr, nullptr);
  run_mlp(m2g_embM, m2g_ef, nullptr, 4, nullptr, nullptr, 0, nullptr, nullptr, 0,
          EM2G, e_m2g, nullptr, nullptr);

  // ---- GNN ----
  gnn_step(g2m_dstM, g2m_edgeM, g2m_srcM, e_g2m, h_grid, h_mesh,
           g2m_src, g2m_dst, EG2M, NM, NG, /*src*/true, /*eadd*/false);
  for (int l = 0; l < NL; ++l)
    gnn_step(m2m_dstM[l], m2m_edgeM[l], m2m_srcM[l], e_m2m, h_mesh, h_mesh,
             m2m_src, m2m_dst, EM2M, NM, NM, true, true);
  gnn_step(m2g_dstM, m2g_edgeM, m2g_srcM, e_m2g, h_mesh, h_grid,
           m2g_src, m2g_dst, EM2G, NG, NM, /*src*/false, /*eadd*/false);

  // ---- decoder ----
  run_mlp(m2g_outM, h_grid, nullptr, 128, nullptr, nullptr, 0, nullptr, nullptr, 0,
          NG, (float*)d_out, nullptr, nullptr);
}
